// MultiHeadSelfAttention_90821378441725
// MI455X (gfx1250) — compile-verified, matrix operand order checked by reading, not run on hardware
//
#include <hip/hip_runtime.h>

// ---------------------------------------------------------------------------
// MI455X (gfx1250) multi-head self-attention, bf16 WMMA + double-buffered
// TDM async staging.
//   x:(2,2048,1024) f32, W_qkv:(3072,1024) f32, W_o:(1024,1024) f32
//   out:(2,2048,1024) f32
// ---------------------------------------------------------------------------

typedef __attribute__((ext_vector_type(4)))  float        v4f;
typedef __attribute__((ext_vector_type(8)))  float        v8f;
typedef __attribute__((ext_vector_type(4)))  __bf16       v4bf;
typedef __attribute__((ext_vector_type(8)))  __bf16       v8bf;
typedef __attribute__((ext_vector_type(16))) __bf16       v16bf;
typedef __attribute__((ext_vector_type(4)))  unsigned int u32x4;
typedef __attribute__((ext_vector_type(8)))  int          i32x8;
typedef __attribute__((ext_vector_type(4)))  int          i32x4;

#define WMMA_BF16(A, B, C) \
    __builtin_amdgcn_wmma_f32_16x16x32_bf16(false, (A), false, (B), (short)0, (C), false, false)

static __device__ __forceinline__ v16bf cat8(v8bf lo, v8bf hi) {
    return __builtin_shufflevector(lo, hi, 0,1,2,3,4,5,6,7,8,9,10,11,12,13,14,15);
}
// Per-lane WMMA 16-bit fragment slice: 8 contiguous bf16 at p, 8 at p+16.
static __device__ __forceinline__ v16bf ldfrag(const __bf16* p) {
    v8bf lo = *(const v8bf*)p;
    v8bf hi = *(const v8bf*)(p + 16);
    return cat8(lo, hi);
}
// Lane broadcast: value from lane `src` (0..31) of this wave.
static __device__ __forceinline__ float lane_bcast(float v, int src) {
    return __builtin_bit_cast(float,
        __builtin_amdgcn_ds_bpermute(src << 2, __builtin_bit_cast(int, v)));
}
// s_wait_tensorcnt needs a literal immediate.
static __device__ __forceinline__ void wait_tensor(bool keep_one) {
    if (keep_one) __builtin_amdgcn_s_wait_tensorcnt(1);
    else          __builtin_amdgcn_s_wait_tensorcnt(0);
}

// ---------------------------------------------------------------------------
// Tensor Data Mover: 2D tile (tile_d0 x tile_d1) of 2-byte elements,
// dim0 contiguous with row stride `stride_elems`, into LDS at lds_off
// (row-major, tightly packed). D# per CDNA5 ISA ch.8; groups 2-3 zero (2D),
// workgroup_mask = 0 (not in a cluster). 6-arg builtin form (clang-23).
// ---------------------------------------------------------------------------
static __device__ __forceinline__ void tdm_load_2d_bf16(unsigned lds_off, const void* gptr,
                                                        unsigned tile_d0, unsigned tile_d1,
                                                        unsigned stride_elems) {
    const unsigned long long ga = (unsigned long long)(uintptr_t)gptr;
    u32x4 g0;
    g0[0] = 1u;                                                // count=1 (valid), load, user
    g0[1] = lds_off;                                           // LDS byte address
    g0[2] = (unsigned)(ga & 0xFFFFFFFFu);                      // global_addr[31:0]
    g0[3] = (unsigned)((ga >> 32) & 0x01FFFFFFu) | (2u << 30); // global_addr[56:32] | type=2
    i32x8 g1;
    g1[0] = (int)(1u << 16);                                   // data_size=1 (2B); wg_mask=0
    g1[1] = (int)((tile_d0 & 0xFFFFu) << 16);                  // tensor_dim0 = tile_d0
    g1[2] = (int)((tile_d0 >> 16) | ((tile_d1 & 0xFFFFu) << 16)); // tensor_dim1 = tile_d1
    g1[3] = (int)((tile_d1 >> 16) | (tile_d0 << 16));          // tile_dim0
    g1[4] = (int)(tile_d1 & 0xFFFFu);                          // tile_dim1 (tile_dim2=0)
    g1[5] = (int)stride_elems;                                 // tensor_dim0_stride[31:0]
    g1[6] = 0;
    g1[7] = 0;
    const i32x4 z4 = {0, 0, 0, 0};
    const i32x8 z8 = {0, 0, 0, 0, 0, 0, 0, 0};
    __builtin_amdgcn_tensor_load_to_lds(g0, g1, z4, z4, z8, 0);
}

// ---------------------------------------------------------------------------
// fp32 -> bf16 elementwise convert (vectorized x4)
// ---------------------------------------------------------------------------
__global__ void cvt_f32_bf16(const float* __restrict__ in, __bf16* __restrict__ out, int n4) {
    int i = blockIdx.x * blockDim.x + threadIdx.x;
    int stride = gridDim.x * blockDim.x;
    for (; i < n4; i += stride) {
        v4f v = ((const v4f*)in)[i];
        v4bf o;
        o[0] = (__bf16)v[0]; o[1] = (__bf16)v[1];
        o[2] = (__bf16)v[2]; o[3] = (__bf16)v[3];
        ((v4bf*)out)[i] = o;
    }
}

// ---------------------------------------------------------------------------
// C[M,N] = A[M,K] @ B[N,K]^T  (row-major, K-contiguous "NT" GEMM)
// 256 thr / 8 waves -> 128x128 tile; wave = 16 rows x 8 N-subtiles.
// B panel (128 x 64) double-buffered in LDS via TDM prefetch (1 iter ahead).
// ---------------------------------------------------------------------------
template <bool F32OUT>
__global__ void gemm_nt_bf16(const __bf16* __restrict__ A, const __bf16* __restrict__ Bm,
                             void* __restrict__ Cv, int M, int N, int K) {
    __shared__ __align__(16) __bf16 Bs[2][128 * 64];   // 2 x 16 KB

    const int lane = threadIdx.x & 31;
    const int wave = threadIdx.x >> 5;
    const int l15  = lane & 15;
    const int kb2  = (lane >> 4) * 8;
    const int n0   = blockIdx.x * 128;
    const int m0   = blockIdx.y * 128 + wave * 16;
    const unsigned bsOff0 = (unsigned)(uintptr_t)(&Bs[0][0]);
    const unsigned bsOff1 = (unsigned)(uintptr_t)(&Bs[1][0]);

    v8f acc[8];
#pragma unroll
    for (int s = 0; s < 8; ++s)
#pragma unroll
        for (int r = 0; r < 8; ++r) acc[s][r] = 0.0f;

    const int nk = K >> 6;                       // 64-wide k-steps
    if (wave == 0)
        tdm_load_2d_bf16(bsOff0, Bm + (size_t)n0 * K, 64u, 128u, (unsigned)K);

    for (int i = 0; i < nk; ++i) {
        const int k0 = i * 64;
        if (wave == 0) {
            if (i + 1 < nk)
                tdm_load_2d_bf16((i & 1) ? bsOff0 : bsOff1,
                                 Bm + (size_t)n0 * K + k0 + 64, 64u, 128u, (unsigned)K);
            wait_tensor(i + 1 < nk);
        }
        __syncthreads();

        const __bf16* Ar = A + (size_t)(m0 + l15) * K + k0 + kb2;
        v16bf af0 = ldfrag(Ar);
        v16bf af1 = ldfrag(Ar + 32);

        const __bf16* bsb = &Bs[i & 1][0];
        // software-pipelined B-fragment reads: next subtile in flight under WMMA
        v16bf b0 = ldfrag(bsb + l15 * 64 + kb2);
        v16bf b1 = ldfrag(bsb + l15 * 64 + kb2 + 32);
#pragma unroll
        for (int s = 0; s < 8; ++s) {
            v16bf nb0 = b0, nb1 = b1;
            if (s < 7) {
                nb0 = ldfrag(bsb + ((s + 1) * 16 + l15) * 64 + kb2);
                nb1 = ldfrag(bsb + ((s + 1) * 16 + l15) * 64 + kb2 + 32);
            }
            acc[s] = WMMA_BF16(af0, b0, acc[s]);
            acc[s] = WMMA_BF16(af1, b1, acc[s]);
            b0 = nb0; b1 = nb1;
        }
        __syncthreads();
    }

    const int rofs = (lane >> 4) * 8;   // C-layout: VGPR r -> row r (+8 for hi lanes)
#pragma unroll
    for (int s = 0; s < 8; ++s) {
#pragma unroll
        for (int r = 0; r < 8; ++r) {
            const size_t m = (size_t)(m0 + r + rofs);
            const size_t n = (size_t)(n0 + s * 16 + l15);
            if (F32OUT) ((float*)Cv)[m * (size_t)N + n] = acc[s][r];
            else        ((__bf16*)Cv)[m * (size_t)N + n] = (__bf16)acc[s][r];
        }
    }
}

// ---------------------------------------------------------------------------
// Causal flash attention. Grid: (S/128, B*H); 8 waves, wave = 16 queries.
// Scores computed TRANSPOSED (S^T = K * Q^T): C-layout has lane=query,
// vgpr=key -> per-lane softmax + one shfl_xor(16); exp'd P tiles are
// bit-identical to the A-fragment layout for O += P*V.
// K block: TDM, double-buffered, prefetched 1 block ahead. V block:
// vector-path transpose (overlaps the in-flight TDM load).
// qkv row = (b*2048+s)*3072; cols [0,1024)=Q [1024,2048)=K [2048,3072)=V.
// ---------------------------------------------------------------------------
__global__ void flash_attn_bf16(const __bf16* __restrict__ qkv, __bf16* __restrict__ hout) {
    __shared__ __align__(16) __bf16 Ks[2][128 * 64];  // key-major, 2 x 16 KB
    __shared__ __align__(16) __bf16 Vt[64 * 128];     // hd-major (transposed), 16 KB

    const int lane = threadIdx.x & 31;
    const int wave = threadIdx.x >> 5;
    const int l15  = lane & 15;
    const int hi8  = (lane >> 4) * 8;
    const int b    = blockIdx.y >> 4;
    const int hh   = blockIdx.y & 15;
    const int qb   = blockIdx.x;
    const int q0w  = qb * 128 + wave * 16;
    const unsigned ksOff0 = (unsigned)(uintptr_t)(&Ks[0][0]);
    const unsigned ksOff1 = (unsigned)(uintptr_t)(&Ks[1][0]);

    const size_t base = (size_t)b * 2048 * 3072 + (size_t)hh * 64;
    const __bf16* Qp = qkv + base;
    const __bf16* Kp = qkv + base + 1024;
    const __bf16* Vp = qkv + base + 2048;

    // Q^T B-fragments, reused for every key block: lane = query, K = hd
    v16bf qf[2];
#pragma unroll
    for (int ks = 0; ks < 2; ++ks)
        qf[ks] = ldfrag(Qp + (size_t)(q0w + l15) * 3072 + ks * 32 + hi8);

    v8f O[4];
#pragma unroll
    for (int ns = 0; ns < 4; ++ns)
#pragma unroll
        for (int r = 0; r < 8; ++r) O[ns][r] = 0.0f;

    float m_run = -3.0e38f;
    float l_run = 0.0f;

    if (wave == 0)
        tdm_load_2d_bf16(ksOff0, Kp, 64u, 128u, 3072u);   // K block 0

    for (int kb = 0; kb <= qb; ++kb) {
        const int k0 = kb * 128;

        // prefetch next K block via TDM; stage V transposed on the vector path
        if (wave == 0 && kb + 1 <= qb)
            tdm_load_2d_bf16((kb & 1) ? ksOff0 : ksOff1,
                             Kp + (size_t)(k0 + 128) * 3072, 64u, 128u, 3072u);
        {
            const int rk = threadIdx.x >> 1;           // key row 0..127
            const int d0 = (threadIdx.x & 1) * 32;     // hd half
            const __bf16* vsrc = Vp + (size_t)(k0 + rk) * 3072 + d0;
            v8bf vv[4];
#pragma unroll
            for (int j = 0; j < 4; ++j) vv[j] = *(const v8bf*)(vsrc + j * 8);
#pragma unroll
            for (int j = 0; j < 32; ++j)
                Vt[(d0 + j) * 128 + rk] = vv[j / 8][j % 8];
        }
        if (wave == 0)
            wait_tensor(kb + 1 <= qb);
        __syncthreads();

        // --- S^T = K * Q^T : 8 key-subtiles x 2 hd-steps, pipelined loads ---
        const __bf16* ksb = &Ks[kb & 1][0];
        float sv[8][8];
        v16bf a0 = ldfrag(ksb + l15 * 64 + hi8);
        v16bf a1 = ldfrag(ksb + l15 * 64 + 32 + hi8);
#pragma unroll
        for (int t = 0; t < 8; ++t) {
            v16bf na0 = a0, na1 = a1;
            if (t < 7) {
                na0 = ldfrag(ksb + ((t + 1) * 16 + l15) * 64 + hi8);
                na1 = ldfrag(ksb + ((t + 1) * 16 + l15) * 64 + 32 + hi8);
            }
            v8f sa;
#pragma unroll
            for (int r = 0; r < 8; ++r) sa[r] = 0.0f;
            sa = WMMA_BF16(a0, qf[0], sa);
            sa = WMMA_BF16(a1, qf[1], sa);
#pragma unroll
            for (int r = 0; r < 8; ++r) sv[t][r] = sa[r] * 0.125f;  // 1/sqrt(64)
            a0 = na0; a1 = na1;
        }

        // causal mask: only the diagonal block needs it
        if (k0 + 127 > q0w) {
            const int q = q0w + l15;
#pragma unroll
            for (int t = 0; t < 8; ++t)
#pragma unroll
                for (int r = 0; r < 8; ++r)
                    if (k0 + t * 16 + r + hi8 > q) sv[t][r] = -3.0e38f;
        }

        // --- online softmax (lane = query; pair (L,L+16) holds disjoint keys) ---
        float mt = sv[0][0];
#pragma unroll
        for (int t = 0; t < 8; ++t)
#pragma unroll
            for (int r = 0; r < 8; ++r) mt = fmaxf(mt, sv[t][r]);
        mt = fmaxf(mt, __shfl_xor(mt, 16));
        const float m_new = fmaxf(m_run, mt);
        const float alpha = __expf(m_run - m_new);

        // fused exp + row-sum + bf16 pack: P lands in WMMA A-fragment layout
        v16bf pf[4];
        float rs = 0.0f;
#pragma unroll
        for (int kk = 0; kk < 4; ++kk) {
            v16bf p;
#pragma unroll
            for (int j = 0; j < 8; ++j) {
                const float p0 = __expf(sv[2 * kk][j]     - m_new);
                const float p1 = __expf(sv[2 * kk + 1][j] - m_new);
                rs += p0 + p1;
                p[j]     = (__bf16)p0;
                p[j + 8] = (__bf16)p1;
            }
            pf[kk] = p;
        }
        rs += __shfl_xor(rs, 16);
        l_run = l_run * alpha + rs;
        m_run = m_new;

        // rescale O: O vgpr r is query (r + hi8); broadcast that query's alpha
#pragma unroll
        for (int r = 0; r < 8; ++r) {
            const float ar = lane_bcast(alpha, r + hi8);
#pragma unroll
            for (int ns = 0; ns < 4; ++ns) O[ns][r] *= ar;
        }

        // --- O += P * V : batch the 4 V-fragments per key-step, then WMMA ---
#pragma unroll
        for (int kk = 0; kk < 4; ++kk) {
            v16bf vf[4];
#pragma unroll
            for (int ns = 0; ns < 4; ++ns)
                vf[ns] = ldfrag(&Vt[(ns * 16 + l15) * 128 + kk * 32 + hi8]);
#pragma unroll
            for (int ns = 0; ns < 4; ++ns)
                O[ns] = WMMA_BF16(pf[kk], vf[ns], O[ns]);
        }

        __syncthreads();   // protect LDS before next block's staging
    }

    // epilogue: O vgpr r is query (r + hi8); its 1/l lives in lane (query)
    const float linv = 1.0f / l_run;
#pragma unroll
    for (int r = 0; r < 8; ++r) {
        const float lr = lane_bcast(linv, r + hi8);
        const int s = q0w + r + hi8;
        __bf16* orow = hout + (size_t)(b * 2048 + s) * 1024 + hh * 64 + l15;
#pragma unroll
        for (int ns = 0; ns < 4; ++ns)
            orow[ns * 16] = (__bf16)(O[ns][r] * lr);
    }
}

// ---------------------------------------------------------------------------
extern "C" void kernel_launch(void* const* d_in, const int* in_sizes, int n_in,
                              void* d_out, int out_size, void* d_ws, size_t ws_size,
                              hipStream_t stream) {
    const float* x    = (const float*)d_in[0];   // (2,2048,1024)
    const float* Wqkv = (const float*)d_in[1];   // (3072,1024)
    const float* Wo   = (const float*)d_in[2];   // (1024,1024)
    float* out = (float*)d_out;

    char* ws = (char*)d_ws;
    __bf16* xb    = (__bf16*)(ws);                        //  8 MiB  4096x1024
    __bf16* wqkvb = (__bf16*)(ws + (size_t)( 8u << 20));  //  6 MiB  3072x1024
    __bf16* wob   = (__bf16*)(ws + (size_t)(14u << 20));  //  2 MiB  1024x1024
    __bf16* qkvb  = (__bf16*)(ws + (size_t)(16u << 20));  // 24 MiB  4096x3072
    __bf16* hb    = (__bf16*)(ws + (size_t)(40u << 20));  //  8 MiB  4096x1024

    cvt_f32_bf16<<<dim3(2048), dim3(256), 0, stream>>>(x,    xb,    (4096 * 1024) / 4);
    cvt_f32_bf16<<<dim3(2048), dim3(256), 0, stream>>>(Wqkv, wqkvb, (3072 * 1024) / 4);
    cvt_f32_bf16<<<dim3(1024), dim3(256), 0, stream>>>(Wo,   wob,   (1024 * 1024) / 4);

    // qkv = x @ Wqkv^T   (M=4096, N=3072, K=1024)
    gemm_nt_bf16<false><<<dim3(24, 32), dim3(256), 0, stream>>>(xb, wqkvb, qkvb, 4096, 3072, 1024);

    // causal flash attention -> h (bf16)
    flash_attn_bf16<<<dim3(16, 32), dim3(256), 0, stream>>>(qkvb, hb);

    // out = h @ Wo^T   (M=4096, N=1024, K=1024), fp32 output
    gemm_nt_bf16<true><<<dim3(8, 32), dim3(256), 0, stream>>>(hb, wob, out, 4096, 1024, 1024);
}
